// MyModel_87522843559163
// MI455X (gfx1250) — matrix operand order repaired and checked by reading, not compiled, not run on hardware
//
#include <hip/hip_runtime.h>
#include <hip/hip_bf16.h>
#include <math.h>

// Problem constants (from the reference)
#define B_   256    // batch
#define H_   2048   // RNN_DIM
#define G_   8192   // 4*H gates
#define SW_  64     // SEQ_WIDTH
#define DIN_ 128    // SEQ_WIDTH + FEAT
#define T_   32     // SEQ_LEN

typedef __bf16 bf16;
typedef __attribute__((ext_vector_type(16))) bf16  v16bf;
typedef __attribute__((ext_vector_type(8)))  bf16  v8bf;
typedef __attribute__((ext_vector_type(8)))  float v8f;
typedef int v4i __attribute__((vector_size(4 * sizeof(int))));
typedef __attribute__((address_space(1))) v4i* gv4i_ptr;   // global
typedef __attribute__((address_space(3))) v4i* lv4i_ptr;   // LDS

__device__ __forceinline__ float sigm(float x) { return 1.0f / (1.0f + __expf(-x)); }

// ---------------------------------------------------------------------------
// gfx1250 async global->LDS copy (ASYNCcnt path), with compile-safe fallback.
// Probe result: param0 is v4i addrspace(1)* (non-const), so signature is
// (global v4i*, lds v4i*, imm offset, imm cpol).
// ---------------------------------------------------------------------------
#if __has_builtin(__builtin_amdgcn_global_load_async_to_lds_b128)
#define ASYNC_LDS_REAL 1
__device__ __forceinline__ void async_cp16(const bf16* g, bf16* l) {
  __builtin_amdgcn_global_load_async_to_lds_b128(
      (gv4i_ptr)(void*)(size_t)(const void*)g,
      (lv4i_ptr)l, 0, 0);
}
#else
#define ASYNC_LDS_REAL 0
__device__ __forceinline__ void async_cp16(const bf16* g, bf16* l) {
  *(v8bf*)l = *(const v8bf*)g;   // synchronous fallback (global load + ds store)
}
#endif

__device__ __forceinline__ void wait_async0() {
#if ASYNC_LDS_REAL
#if __has_builtin(__builtin_amdgcn_s_wait_asynccnt)
  __builtin_amdgcn_s_wait_asynccnt(0);
#else
  asm volatile("s_wait_asynccnt 0x0" ::: "memory");
#endif
#endif
}

// ---------------------------------------------------------------------------
// Weight pre-pass: fp32 [K,N] -> bf16 [N,K]  (B-fragment friendly layout).
// 129 MB total after conversion -> resident in MI455X's 192 MB L2.
// ---------------------------------------------------------------------------
__global__ __launch_bounds__(256) void k_cvt_transpose(const float* __restrict__ W,
                                                       bf16* __restrict__ Wt,
                                                       int K, int N) {
  size_t total = (size_t)K * (size_t)N;
  for (size_t i = (size_t)blockIdx.x * blockDim.x + threadIdx.x; i < total;
       i += (size_t)gridDim.x * blockDim.x) {
    size_t k = i % (size_t)K;
    size_t n = i / (size_t)K;
    Wt[i] = (bf16)W[k * (size_t)N + n];   // writes contiguous in k
  }
}

// prev = concat(rand_prev, xs) as bf16 [B, 128]; xs half written once.
__global__ __launch_bounds__(256) void k_init_prev(const float* __restrict__ rp,
                                                   const float* __restrict__ x,
                                                   bf16* __restrict__ prevb) {
  const int i = blockIdx.x * 256 + threadIdx.x;     // B*DIN
  const int b = i >> 7;
  const int j = i & 127;
  const float v = (j < SW_) ? rp[b * SW_ + j] : x[b * SW_ + (j - SW_)];
  prevb[i] = (bf16)v;
}

// State copies: never mutate d_in. h -> bf16 (GEMM operand), c -> fp32.
__global__ __launch_bounds__(256) void k_init_state(const float* __restrict__ h0,
                                                    const float* __restrict__ c0,
                                                    const float* __restrict__ h1,
                                                    const float* __restrict__ c1,
                                                    bf16* __restrict__ h0b, float* __restrict__ c0w,
                                                    bf16* __restrict__ h1b, float* __restrict__ c1w) {
  const int i = blockIdx.x * 256 + threadIdx.x;     // B*H
  h0b[i] = (bf16)h0[i]; c0w[i] = c0[i];
  h1b[i] = (bf16)h1[i]; c1w[i] = c1[i];
}

// ---------------------------------------------------------------------------
// WMMA fragment loaders, matching CDNA5 ISA 16-bit layouts.
// A (16x32, row-major bf16): lane m=lane&15, kg=lane>>4,
//   elements 0..7 -> K = k0+kg*8+e ; elements 8..15 -> K = k0+16+kg*8+(e-8)
// B (32x16 from [N,K]-packed tile): lane n=lane&15, kg=lane>>4,
//   same K mapping as A: e<8 -> K = kg*8+e ; e>=8 -> K = 16+kg*8+(e-8)
// ---------------------------------------------------------------------------
__device__ __forceinline__ v16bf load_afrag(const bf16* __restrict__ A, int lda,
                                            int m_base, int k0, int lane) {
  const int m = lane & 15, kg = lane >> 4;
  const bf16* p = A + (size_t)(m_base + m) * lda + k0 + kg * 8;
  v8bf a0 = *(const v8bf*)(p);
  v8bf a1 = *(const v8bf*)(p + 16);
  v16bf a;
#pragma unroll
  for (int i = 0; i < 8; ++i) { a[i] = a0[i]; a[i + 8] = a1[i]; }
  return a;
}

// B fragment from an LDS tile stored as 64 rows (n) x 32 bf16 (k).
__device__ __forceinline__ v16bf load_bfrag_lds(const bf16* Ls, int t, int lane) {
  const int n = lane & 15, kg = lane >> 4;
  const bf16* p = Ls + (t * 16 + n) * 32 + kg * 8;
  v8bf b0 = *(const v8bf*)(p);
  v8bf b1 = *(const v8bf*)(p + 16);
  v16bf b;
#pragma unroll
  for (int i = 0; i < 8; ++i) { b[i] = b0[i]; b[i + 8] = b1[i]; }
  return b;
}

// ---------------------------------------------------------------------------
// GEMM: C[M,N] = A1*B1t^T (+ A2*B2t^T) + bias, bf16 inputs, fp32 accum.
// Block = 256 threads = 8 wave32s. Wave w -> rows [by*128 + w*16, +16),
// cols [bx*64, +64) as 4 accumulators. B tiles staged once per block into
// LDS via async global->LDS copies (double-buffered, overlapped with WMMA);
// all 8 waves share the staged tile => 8x less weight traffic from L2.
// DUAL fuses x*W + h*U into one accumulation pass (halves z traffic).
// ---------------------------------------------------------------------------
template <bool DUAL, bool OUT_BF16>
__global__ __launch_bounds__(256) void k_wmma_gemm(const bf16* __restrict__ A1,
                                                   const bf16* __restrict__ B1t,
                                                   const bf16* __restrict__ A2,
                                                   const bf16* __restrict__ B2t,
                                                   const float* __restrict__ bias,
                                                   float* __restrict__ Cf,
                                                   bf16* __restrict__ Cb,
                                                   int lda, int ldb, int ldc, int K) {
  constexpr int NMAT = DUAL ? 2 : 1;
  __shared__ bf16 smem[2][NMAT][64 * 32];   // [buf][matrix][n x k] : 8/16 KB

  const int lane = threadIdx.x & 31;
  const int wave = threadIdx.x >> 5;
  const int m_base = blockIdx.y * 128 + wave * 16;
  const int n_base = blockIdx.x * 64;

  // Staging coords: 64 rows x 64B; 256 threads move one 16B chunk each.
  const int srow = threadIdx.x >> 2;        // 0..63
  const int schk = (threadIdx.x & 3) * 8;   // bf16 offset of 16B chunk

  v8f acc[4];
#pragma unroll
  for (int t = 0; t < 4; ++t)
#pragma unroll
    for (int r = 0; r < 8; ++r) acc[t][r] = 0.0f;

  // Prologue: stage first K-chunk into buffer 0.
  {
    async_cp16(B1t + (size_t)(n_base + srow) * ldb + schk,
               &smem[0][0][srow * 32 + schk]);
    if constexpr (DUAL)
      async_cp16(B2t + (size_t)(n_base + srow) * ldb + schk,
                 &smem[0][NMAT - 1][srow * 32 + schk]);
  }

  int buf = 0;
  for (int k0 = 0; k0 < K; k0 += 32) {
    wait_async0();
    __syncthreads();                          // current buffer ready

    if (k0 + 32 < K) {                        // overlap next copy with compute
      async_cp16(B1t + (size_t)(n_base + srow) * ldb + (k0 + 32) + schk,
                 &smem[buf ^ 1][0][srow * 32 + schk]);
      if constexpr (DUAL)
        async_cp16(B2t + (size_t)(n_base + srow) * ldb + (k0 + 32) + schk,
                   &smem[buf ^ 1][NMAT - 1][srow * 32 + schk]);
      __builtin_prefetch((const void*)(A1 + (size_t)(m_base + (lane & 15)) * lda + k0 + 32), 0, 1);
    }

    v16bf a1 = load_afrag(A1, lda, m_base, k0, lane);
#pragma unroll
    for (int t = 0; t < 4; ++t) {
      v16bf b = load_bfrag_lds(&smem[buf][0][0], t, lane);
      acc[t] = __builtin_amdgcn_wmma_f32_16x16x32_bf16(
          false, a1, false, b, (short)0, acc[t], false, false);
    }
    if constexpr (DUAL) {
      v16bf a2 = load_afrag(A2, lda, m_base, k0, lane);
#pragma unroll
      for (int t = 0; t < 4; ++t) {
        v16bf b = load_bfrag_lds(&smem[buf][NMAT - 1][0], t, lane);
        acc[t] = __builtin_amdgcn_wmma_f32_16x16x32_bf16(
            false, a2, false, b, (short)0, acc[t], false, false);
      }
    }

    __syncthreads();                          // reads done before buf restaged
    buf ^= 1;
  }

  // C/D layout: VGPR r -> M = r + 8*(lane>>4), N = lane&15
  const int n = lane & 15;
  const int mo = (lane >> 4) * 8;
#pragma unroll
  for (int t = 0; t < 4; ++t) {
    const int col = n_base + t * 16 + n;
    const float bv = bias[col];
#pragma unroll
    for (int r = 0; r < 8; ++r) {
      const int row = m_base + mo + r;
      const float v = acc[t][r] + bv;
      if constexpr (OUT_BF16) Cb[(size_t)row * ldc + col] = (bf16)v;
      else                    Cf[(size_t)row * ldc + col] = v;
    }
  }
}

// LSTM gate nonlinearity. Keras order i,f,g,o. c fp32 state, h -> bf16.
__global__ __launch_bounds__(256) void k_lstm_pointwise(const float* __restrict__ z,
                                                        float* __restrict__ c,
                                                        bf16* __restrict__ hb) {
  const int i = blockIdx.x * 256 + threadIdx.x;   // B*H
  const int b = i >> 11;                          // / 2048
  const int j = i & (H_ - 1);
  const float* zr = z + (size_t)b * (4 * H_);
  const float zi = zr[j];
  const float zf = zr[H_ + j];
  const float zg = zr[2 * H_ + j];
  const float zo = zr[3 * H_ + j];
  const float cn = sigm(zf) * c[i] + sigm(zi) * tanhf(zg);
  c[i] = cn;
  hb[i] = (bf16)(sigm(zo) * tanhf(cn));
}

// out[b,t,:] = sigmoid(zo); rebuild prev (first 64 cols) for the next step.
__global__ __launch_bounds__(256) void k_out_pointwise(const float* __restrict__ zo,
                                                       float* __restrict__ out,
                                                       bf16* __restrict__ prevb,
                                                       int t) {
  const int i = blockIdx.x * 256 + threadIdx.x;   // B*SW
  const int b = i >> 6;
  const int f = i & 63;
  const float v = sigm(zo[i]);
  out[(size_t)b * (T_ * SW_) + (size_t)t * SW_ + f] = v;
  prevb[b * DIN_ + f] = (bf16)v;
}

// ---------------------------------------------------------------------------
extern "C" void kernel_launch(void* const* d_in, const int* in_sizes, int n_in,
                              void* d_out, int out_size, void* d_ws, size_t ws_size,
                              hipStream_t stream) {
  (void)in_sizes; (void)n_in; (void)out_size; (void)ws_size;
  const float* x     = (const float*)d_in[0];
  const float* rp    = (const float*)d_in[1];
  const float* h0_0  = (const float*)d_in[2];
  const float* c0_0  = (const float*)d_in[3];
  const float* h0_1  = (const float*)d_in[4];
  const float* c0_1  = (const float*)d_in[5];
  const float* W_in  = (const float*)d_in[6];
  const float* b_in  = (const float*)d_in[7];
  const float* Wk0   = (const float*)d_in[8];
  const float* Uk0   = (const float*)d_in[9];
  const float* bk0   = (const float*)d_in[10];
  const float* Wk1   = (const float*)d_in[11];
  const float* Uk1   = (const float*)d_in[12];
  const float* bk1   = (const float*)d_in[13];
  const float* W_out = (const float*)d_in[14];
  const float* b_out = (const float*)d_in[15];
  float* out = (float*)d_out;

  // Workspace carve (~146 MB): bf16 weights (129 MB, L2-resident) + activations.
  char* p = (char*)d_ws;
  auto carve = [&](size_t bytes) -> char* {
    char* r = p; p += (bytes + 255) & ~(size_t)255; return r;
  };
  bf16* Wint  = (bf16*)carve((size_t)H_  * DIN_ * 2);  // [H,  DIN]
  bf16* Wk0t  = (bf16*)carve((size_t)G_  * H_   * 2);  // [G,  H]
  bf16* Uk0t  = (bf16*)carve((size_t)G_  * H_   * 2);
  bf16* Wk1t  = (bf16*)carve((size_t)G_  * H_   * 2);
  bf16* Uk1t  = (bf16*)carve((size_t)G_  * H_   * 2);
  bf16* Woutt = (bf16*)carve((size_t)SW_ * H_   * 2);  // [SW, H]
  bf16* prevb = (bf16*)carve((size_t)B_ * DIN_ * 2);
  bf16* xbuf  = (bf16*)carve((size_t)B_ * H_   * 2);
  bf16* h0b   = (bf16*)carve((size_t)B_ * H_   * 2);
  bf16* h1b   = (bf16*)carve((size_t)B_ * H_   * 2);
  float* c0w  = (float*)carve((size_t)B_ * H_  * 4);
  float* c1w  = (float*)carve((size_t)B_ * H_  * 4);
  float* z    = (float*)carve((size_t)B_ * G_  * 4);
  float* zo   = (float*)carve((size_t)B_ * SW_ * 4);

  // One-time weight convert+pack (deterministic, done every call).
  k_cvt_transpose<<<1024, 256, 0, stream>>>(W_in,  Wint,  DIN_, H_);
  k_cvt_transpose<<<8192, 256, 0, stream>>>(Wk0,   Wk0t,  H_,   G_);
  k_cvt_transpose<<<8192, 256, 0, stream>>>(Uk0,   Uk0t,  H_,   G_);
  k_cvt_transpose<<<8192, 256, 0, stream>>>(Wk1,   Wk1t,  H_,   G_);
  k_cvt_transpose<<<8192, 256, 0, stream>>>(Uk1,   Uk1t,  H_,   G_);
  k_cvt_transpose<<<1024, 256, 0, stream>>>(W_out, Woutt, H_,   SW_);
  k_init_prev<<<(B_ * DIN_) / 256, 256, 0, stream>>>(rp, x, prevb);
  k_init_state<<<(B_ * H_) / 256, 256, 0, stream>>>(h0_0, c0_0, h0_1, c0_1,
                                                    h0b, c0w, h1b, c1w);

  const dim3 blk(256);
  const dim3 gIn(H_ / 64, B_ / 128);   // (32, 2)
  const dim3 gG (G_ / 64, B_ / 128);   // (128, 2)
  const dim3 gO (SW_ / 64, B_ / 128);  // (1, 2)

  for (int t = 0; t < T_; ++t) {
    // cell_in = prev @ W_in + b_in  -> bf16
    hipLaunchKernelGGL(HIP_KERNEL_NAME(k_wmma_gemm<false, true>), gIn, blk, 0, stream,
                       prevb, Wint, (const bf16*)nullptr, (const bf16*)nullptr, b_in,
                       (float*)nullptr, xbuf, DIN_, DIN_, H_, DIN_);
    // z0 = cell_in @ Wk0 + h0 @ Uk0 + bk0   (fused dual GEMM)
    hipLaunchKernelGGL(HIP_KERNEL_NAME(k_wmma_gemm<true, false>), gG, blk, 0, stream,
                       (const bf16*)xbuf, Wk0t, (const bf16*)h0b, Uk0t, bk0,
                       z, (bf16*)nullptr, H_, H_, G_, H_);
    k_lstm_pointwise<<<(B_ * H_) / 256, 256, 0, stream>>>(z, c0w, h0b);
    // z1 = h0 @ Wk1 + h1 @ Uk1 + bk1
    hipLaunchKernelGGL(HIP_KERNEL_NAME(k_wmma_gemm<true, false>), gG, blk, 0, stream,
                       (const bf16*)h0b, Wk1t, (const bf16*)h1b, Uk1t, bk1,
                       z, (bf16*)nullptr, H_, H_, G_, H_);
    k_lstm_pointwise<<<(B_ * H_) / 256, 256, 0, stream>>>(z, c1w, h1b);
    // zo = h1 @ W_out + b_out
    hipLaunchKernelGGL(HIP_KERNEL_NAME(k_wmma_gemm<false, false>), gO, blk, 0, stream,
                       (const bf16*)h1b, Woutt, (const bf16*)nullptr, (const bf16*)nullptr,
                       b_out, zo, (bf16*)nullptr, H_, H_, SW_, H_);
    k_out_pointwise<<<(B_ * SW_) / 256, 256, 0, stream>>>(zo, out, prevb, t);
  }
}
